// StructPool_65223373357289
// MI455X (gfx1250) — 16-bit and 8-bit matrix operand order repaired and checked by reading, compiled, not run on hardware
//
#include <hip/hip_runtime.h>
#include <hip/hip_bf16.h>

#define B_   32
#define N_   1024
#define FIN_ 128
#define H_   64
#define C_   32
#define KG_  2
#define NUM_ITER_ 5
#define INV_TEMP_ 2.0f

typedef __attribute__((ext_vector_type(16))) _Float16 v16h;
typedef __attribute__((ext_vector_type(8)))  float    v8f;
typedef __attribute__((ext_vector_type(8)))  int      v8i;

__device__ __forceinline__ v16h frag_a_f16(const _Float16* rowp, int kb, int lane) {
  const _Float16* p = rowp + kb + ((lane >> 4) << 3);
  union { v16h v; uint4 q[2]; } u;
  u.q[0] = *(const uint4*)p;
  u.q[1] = *(const uint4*)(p + 16);
  return u.v;
}
__device__ __forceinline__ v16h frag_b_f16(const _Float16* colp, int kb, int lane) {
  const _Float16* p = colp + kb + ((lane >> 4) << 3);
  union { v16h v; uint4 q[2]; } u;
  u.q[0] = *(const uint4*)p;
  u.q[1] = *(const uint4*)(p + 16);
  return u.v;
}
__device__ __forceinline__ unsigned h2_from01(unsigned w) {
  return ((w & 1u) | ((w & 0x100u) << 8)) * 0x3C00u;
}
__device__ __forceinline__ v16h frag_a_u8_to_f16(const unsigned char* rowp, int kb, int lane) {
  const unsigned* p0 = (const unsigned*)(rowp + kb + ((lane >> 4) << 3));
  const unsigned* p1 = (const unsigned*)(rowp + kb + ((lane >> 4) << 3) + 16);
  unsigned w0 = p0[0], w1 = p0[1], w2 = p1[0], w3 = p1[1];
  union { v16h v; unsigned d[8]; } u;
  u.d[0] = h2_from01(w0);
  u.d[1] = h2_from01(w0 >> 16);
  u.d[2] = h2_from01(w1);
  u.d[3] = h2_from01(w1 >> 16);
  u.d[4] = h2_from01(w2);
  u.d[5] = h2_from01(w2 >> 16);
  u.d[6] = h2_from01(w3);
  u.d[7] = h2_from01(w3 >> 16);
  return u.v;
}
__device__ __forceinline__ v8i frag_a_iu8(const unsigned char* rowp, int kb, int lane) {
  const unsigned char* p = rowp + kb + ((lane >> 4) << 3);
  union { v8i v; uint2 d[4]; } u;
  u.d[0] = *(const uint2*)(p);
  u.d[1] = *(const uint2*)(p + 16);
  u.d[2] = *(const uint2*)(p + 32);
  u.d[3] = *(const uint2*)(p + 48);
  return u.v;
}
__device__ __forceinline__ v8i frag_b_iu8(const unsigned char* colp, int kb, int lane) {
  const unsigned char* p = colp + kb + ((lane >> 4) << 3);
  union { v8i v; uint2 d[4]; } u;
  u.d[0] = *(const uint2*)(p);
  u.d[1] = *(const uint2*)(p + 16);
  u.d[2] = *(const uint2*)(p + 32);
  u.d[3] = *(const uint2*)(p + 48);
  return u.v;
}

__global__ void k_prep(const int* __restrict__ adj, unsigned char* __restrict__ adj8,
                       unsigned char* __restrict__ adjT8, float* __restrict__ dinv) {
  int bm = blockIdx.x;
  int b = bm >> 10, m = bm & (N_ - 1);
  const int* row = adj + (size_t)bm * N_;
  unsigned char* o  = adj8  + (size_t)bm * N_;
  unsigned char* ot = adjT8 + (size_t)b * N_ * N_ + m;
  int t = threadIdx.x;
  int s = 0;
  for (int j = t; j < N_; j += 256) {
    unsigned char v = (unsigned char)(row[j] != 0 ? 1 : 0);
    o[j] = v;
    ot[(size_t)j * N_] = v;
    s += v;
  }
  __shared__ int red[256];
  red[t] = s; __syncthreads();
  for (int k = 128; k > 0; k >>= 1) { if (t < k) red[t] += red[t + k]; __syncthreads(); }
  if (t == 0) dinv[bm] = rsqrtf(1.0f + (float)red[0]);
}

__global__ void k_xw(const float* __restrict__ x, const float* __restrict__ Wg,
                     const float* __restrict__ dinv, _Float16* __restrict__ ysT) {
  int bn = blockIdx.x;
  int b = bn >> 10, n = bn & (N_ - 1);
  __shared__ float xr[FIN_];
  int t = threadIdx.x;
  for (int f = t; f < FIN_; f += H_) xr[f] = x[(size_t)bn * FIN_ + f];
  __syncthreads();
  float s = 0.f;
  for (int f = 0; f < FIN_; ++f) s += xr[f] * Wg[f * H_ + t];
  ysT[((size_t)b * H_ + t) * N_ + n] = (_Float16)(s * dinv[bn]);
}

__global__ void k_gcn(const unsigned char* __restrict__ adj8, const _Float16* __restrict__ ysT,
                      const float* __restrict__ dinv, const float* __restrict__ bg,
                      float* __restrict__ h, _Float16* __restrict__ hT) {
  int b  = blockIdx.y;
  int Mt = blockIdx.x * 16;
  int lane = threadIdx.x & 31;
  int l15 = lane & 15;
  int moff = (lane >> 4) << 3;
  const unsigned char* Arow = adj8 + ((size_t)b * N_ + Mt + l15) * N_;
  v8f acc[4] = {};
  for (int kb = 0; kb < N_; kb += 32) {
    __builtin_prefetch(Arow + kb + 32, 0, 1);
    v16h af = frag_a_u8_to_f16(Arow, kb, lane);
#pragma unroll
    for (int t = 0; t < 4; ++t) {
      v16h bf = frag_b_f16(ysT + ((size_t)b * H_ + t * 16 + l15) * N_, kb, lane);
      acc[t] = __builtin_amdgcn_wmma_f32_16x16x32_f16(false, af, false, bf,
                                                      (short)0, acc[t], false, false);
    }
  }
  int m0 = Mt + moff;
  float dvv[8];
#pragma unroll
  for (int r = 0; r < 8; ++r) dvv[r] = dinv[(size_t)b * N_ + m0 + r];
#pragma unroll
  for (int t = 0; t < 4; ++t) {
    int hc = t * 16 + l15;
    const _Float16* yself = ysT + ((size_t)b * H_ + hc) * N_ + m0;
    float bias = bg[hc];
    alignas(16) _Float16 pk[8];
#pragma unroll
    for (int r = 0; r < 8; ++r) {
      float v = dvv[r] * (acc[t][r] + (float)yself[r]) + bias;
      v = v > 0.f ? v : 0.f;
      h[((size_t)b * N_ + m0 + r) * H_ + hc] = v;
      pk[r] = (_Float16)v;
    }
    *(uint4*)(hT + ((size_t)b * H_ + hc) * N_ + m0) = *(const uint4*)pk;
  }
}

__global__ void k_eu(const float* __restrict__ h, const float* __restrict__ Wu,
                     const float* __restrict__ bu, const float* __restrict__ means,
                     const float* __restrict__ scales, float* __restrict__ e,
                     float* __restrict__ unary) {
  __shared__ float sW[H_ * C_];
  __shared__ float sm[KG_ * H_], ss[KG_ * H_];
  int t = threadIdx.x;
  for (int i = t; i < H_ * C_; i += 256) sW[i] = Wu[i];
  for (int i = t; i < KG_ * H_; i += 256) { sm[i] = means[i]; ss[i] = scales[i]; }
  __syncthreads();
  int bn = blockIdx.x * 256 + t;
  const float* hr = h + (size_t)bn * H_;
  float hl[H_];
#pragma unroll
  for (int i = 0; i < H_; ++i) hl[i] = hr[i];
#pragma unroll
  for (int k = 0; k < KG_; ++k) {
    float d = 0.f;
#pragma unroll
    for (int i = 0; i < H_; ++i) {
      float df = (hl[i] - sm[k * H_ + i]) / ss[k * H_ + i];
      d += df * df;
    }
    e[(size_t)bn * KG_ + k] = expf(-d);
  }
  for (int c = 0; c < C_; ++c) {
    float s = bu[c];
#pragma unroll
    for (int i = 0; i < H_; ++i) s += hl[i] * sW[i * C_ + c];
    unary[(size_t)bn * C_ + c] = s;
  }
}

__global__ void k_pair(const unsigned char* __restrict__ adj8,
                       const unsigned char* __restrict__ adjT8,
                       const float* __restrict__ e, const float* __restrict__ kw,
                       _Float16* __restrict__ P) {
  int b  = blockIdx.z;
  int Mt = blockIdx.x * 16;
  int Jt = blockIdx.y * 64;
  int lane = threadIdx.x & 31;
  int l15 = lane & 15;
  int moff = (lane >> 4) << 3;
  const unsigned char* Arow = adj8 + ((size_t)b * N_ + Mt + l15) * N_;
  v8i acc[4] = {};
  for (int kb = 0; kb < N_; kb += 64) {
    __builtin_prefetch(Arow + kb + 64, 0, 1);
    v8i af = frag_a_iu8(Arow, kb, lane);
#pragma unroll
    for (int t = 0; t < 4; ++t) {
      v8i bf = frag_b_iu8(adjT8 + ((size_t)b * N_ + Jt + t * 16 + l15) * N_, kb, lane);
      acc[t] = __builtin_amdgcn_wmma_i32_16x16x64_iu8(false, af, false, bf,
                                                      acc[t], false, false);
    }
  }
  float kw0 = kw[0], kw1 = kw[1];
  int m0 = Mt + moff;
  float em0[8], em1[8];
#pragma unroll
  for (int r = 0; r < 8; ++r) {
    em0[r] = e[((size_t)b * N_ + m0 + r) * KG_ + 0];
    em1[r] = e[((size_t)b * N_ + m0 + r) * KG_ + 1];
  }
#pragma unroll
  for (int t = 0; t < 4; ++t) {
    int j = Jt + t * 16 + l15;
    float ej0 = e[((size_t)b * N_ + j) * KG_ + 0];
    float ej1 = e[((size_t)b * N_ + j) * KG_ + 1];
#pragma unroll
    for (int r = 0; r < 8; ++r) {
      float pv = (acc[t][r] > 0) ? (kw0 * em0[r] * ej0 + kw1 * em1[r] * ej1) : 0.f;
      P[((size_t)b * N_ + m0 + r) * N_ + j] = (_Float16)pv;
    }
  }
}

__global__ void k_soft(const float* __restrict__ unary, const float* __restrict__ msg,
                       const float* __restrict__ mu, int use_msg, float inv_temp,
                       _Float16* __restrict__ QT, _Float16* __restrict__ QMT) {
  __shared__ float smu[C_ * C_];
  int t = threadIdx.x;
  for (int i = t; i < C_ * C_; i += 256) smu[i] = mu[i];
  __syncthreads();
  int bn = blockIdx.x * 256 + t;
  int b = bn >> 10, n = bn & (N_ - 1);
  float q[C_];
  float mx = -1e30f;
#pragma unroll
  for (int c = 0; c < C_; ++c) {
    float v = unary[(size_t)bn * C_ + c];
    if (use_msg) v += msg[(size_t)bn * C_ + c];
    v *= inv_temp;
    q[c] = v;
    mx = fmaxf(mx, v);
  }
  float s = 0.f;
#pragma unroll
  for (int c = 0; c < C_; ++c) { q[c] = expf(q[c] - mx); s += q[c]; }
  float inv = 1.0f / s;
#pragma unroll
  for (int c = 0; c < C_; ++c) {
    q[c] *= inv;
    QT[((size_t)b * C_ + c) * N_ + n] = (_Float16)q[c];
  }
  for (int c2 = 0; c2 < C_; ++c2) {
    float s2 = 0.f;
#pragma unroll
    for (int c = 0; c < C_; ++c) s2 += q[c] * smu[c * C_ + c2];
    QMT[((size_t)b * C_ + c2) * N_ + n] = (_Float16)s2;
  }
}

__global__ void k_msg(const _Float16* __restrict__ P, const _Float16* __restrict__ QMT,
                      float* __restrict__ msg) {
  int b  = blockIdx.y;
  int Mt = blockIdx.x * 16;
  int lane = threadIdx.x & 31;
  int l15 = lane & 15;
  int moff = (lane >> 4) << 3;
  const _Float16* Arow = P + ((size_t)b * N_ + Mt + l15) * N_;
  v8f acc[2] = {};
  for (int kb = 0; kb < N_; kb += 32) {
    __builtin_prefetch(Arow + kb + 32, 0, 1);
    v16h af = frag_a_f16(Arow, kb, lane);
#pragma unroll
    for (int t = 0; t < 2; ++t) {
      v16h bf = frag_b_f16(QMT + ((size_t)b * C_ + t * 16 + l15) * N_, kb, lane);
      acc[t] = __builtin_amdgcn_wmma_f32_16x16x32_f16(false, af, false, bf,
                                                      (short)0, acc[t], false, false);
    }
  }
  int m0 = Mt + moff;
#pragma unroll
  for (int t = 0; t < 2; ++t)
#pragma unroll
    for (int r = 0; r < 8; ++r)
      msg[((size_t)b * N_ + m0 + r) * C_ + t * 16 + l15] = acc[t][r];
}

__global__ void k_xpool(const _Float16* __restrict__ QT, const _Float16* __restrict__ hT,
                        float* __restrict__ outx) {
  int b  = blockIdx.z;
  int Mt = blockIdx.x * 16;
  int Nt = blockIdx.y * 16;
  int lane = threadIdx.x & 31;
  int l15 = lane & 15;
  int moff = (lane >> 4) << 3;
  const _Float16* Arow = QT + ((size_t)b * C_ + Mt + l15) * N_;
  v8f acc = {};
  for (int kb = 0; kb < N_; kb += 32) {
    v16h af = frag_a_f16(Arow, kb, lane);
    v16h bf = frag_b_f16(hT + ((size_t)b * H_ + Nt + l15) * N_, kb, lane);
    acc = __builtin_amdgcn_wmma_f32_16x16x32_f16(false, af, false, bf,
                                                 (short)0, acc, false, false);
  }
#pragma unroll
  for (int r = 0; r < 8; ++r)
    outx[((size_t)b * C_ + Mt + moff + r) * H_ + Nt + l15] = acc[r];
}

__global__ void k_aq(const unsigned char* __restrict__ adj8, const _Float16* __restrict__ QT,
                     _Float16* __restrict__ T1T) {
  int b  = blockIdx.z;
  int Mt = blockIdx.x * 16;
  int ct = blockIdx.y;
  int lane = threadIdx.x & 31;
  int l15 = lane & 15;
  int moff = (lane >> 4) << 3;
  const unsigned char* Arow = adj8 + ((size_t)b * N_ + Mt + l15) * N_;
  v8f acc = {};
  for (int kb = 0; kb < N_; kb += 32) {
    __builtin_prefetch(Arow + kb + 32, 0, 1);
    v16h af = frag_a_u8_to_f16(Arow, kb, lane);
    v16h bf = frag_b_f16(QT + ((size_t)b * C_ + ct * 16 + l15) * N_, kb, lane);
    acc = __builtin_amdgcn_wmma_f32_16x16x32_f16(false, af, false, bf,
                                                 (short)0, acc, false, false);
  }
  alignas(16) _Float16 pk[8];
#pragma unroll
  for (int r = 0; r < 8; ++r) pk[r] = (_Float16)acc[r];
  *(uint4*)(T1T + ((size_t)b * C_ + ct * 16 + l15) * N_ + Mt + moff) = *(const uint4*)pk;
}

__global__ void k_apool(const _Float16* __restrict__ QT, const _Float16* __restrict__ T1T,
                        float* __restrict__ outa) {
  int b  = blockIdx.z;
  int Mt = blockIdx.x * 16;
  int Nt = blockIdx.y * 16;
  int lane = threadIdx.x & 31;
  int l15 = lane & 15;
  int moff = (lane >> 4) << 3;
  const _Float16* Arow = QT + ((size_t)b * C_ + Mt + l15) * N_;
  v8f acc = {};
  for (int kb = 0; kb < N_; kb += 32) {
    v16h af = frag_a_f16(Arow, kb, lane);
    v16h bf = frag_b_f16(T1T + ((size_t)b * C_ + Nt + l15) * N_, kb, lane);
    acc = __builtin_amdgcn_wmma_f32_16x16x32_f16(false, af, false, bf,
                                                 (short)0, acc, false, false);
  }
#pragma unroll
  for (int r = 0; r < 8; ++r)
    outa[((size_t)b * C_ + Mt + moff + r) * C_ + Nt + l15] = acc[r];
}

extern "C" void kernel_launch(void* const* d_in, const int* in_sizes, int n_in,
                              void* d_out, int out_size, void* d_ws, size_t ws_size,
                              hipStream_t stream) {
  const float* x      = (const float*)d_in[0];
  const int*   adj    = (const int*)  d_in[1];
  const float* W_gcn  = (const float*)d_in[2];
  const float* b_gcn  = (const float*)d_in[3];
  const float* W_un   = (const float*)d_in[4];
  const float* b_un   = (const float*)d_in[5];
  const float* means  = (const float*)d_in[6];
  const float* scales = (const float*)d_in[7];
  const float* kw     = (const float*)d_in[8];
  const float* mu     = (const float*)d_in[9];
  float* out = (float*)d_out;
  (void)in_sizes; (void)n_in; (void)out_size; (void)ws_size;

  char* w = (char*)d_ws;
  auto alloc = [&](size_t bytes) -> void* {
    void* p = (void*)w;
    w += (bytes + 255) & ~(size_t)255;
    return p;
  };
  const size_t NN = (size_t)B_ * N_ * N_;
  unsigned char* adj8  = (unsigned char*)alloc(NN);
  unsigned char* adjT8 = (unsigned char*)alloc(NN);
  float*     dinv  = (float*)    alloc((size_t)B_ * N_ * 4);
  _Float16*  ysT   = (_Float16*) alloc((size_t)B_ * H_ * N_ * 2);
  float*     h     = (float*)    alloc((size_t)B_ * N_ * H_ * 4);
  _Float16*  hT    = (_Float16*) alloc((size_t)B_ * H_ * N_ * 2);
  float*     e     = (float*)    alloc((size_t)B_ * N_ * KG_ * 4);
  float*     unary = (float*)    alloc((size_t)B_ * N_ * C_ * 4);
  _Float16*  P     = (_Float16*) alloc(NN * 2);
  _Float16*  QT    = (_Float16*) alloc((size_t)B_ * C_ * N_ * 2);
  _Float16*  QMT   = (_Float16*) alloc((size_t)B_ * C_ * N_ * 2);
  float*     msg   = (float*)    alloc((size_t)B_ * N_ * C_ * 4);
  _Float16*  T1T   = (_Float16*) alloc((size_t)B_ * C_ * N_ * 2);

  k_prep<<<B_ * N_, 256, 0, stream>>>(adj, adj8, adjT8, dinv);
  k_xw<<<B_ * N_, H_, 0, stream>>>(x, W_gcn, dinv, ysT);
  k_gcn<<<dim3(N_ / 16, B_), 32, 0, stream>>>(adj8, ysT, dinv, b_gcn, h, hT);
  k_eu<<<(B_ * N_) / 256, 256, 0, stream>>>(h, W_un, b_un, means, scales, e, unary);
  k_pair<<<dim3(N_ / 16, N_ / 64, B_), 32, 0, stream>>>(adj8, adjT8, e, kw, P);
  k_soft<<<(B_ * N_) / 256, 256, 0, stream>>>(unary, msg, mu, 0, 1.0f, QT, QMT);
  for (int it = 0; it < NUM_ITER_; ++it) {
    k_msg<<<dim3(N_ / 16, B_), 32, 0, stream>>>(P, QMT, msg);
    k_soft<<<(B_ * N_) / 256, 256, 0, stream>>>(unary, msg, mu, 1, INV_TEMP_, QT, QMT);
  }
  k_xpool<<<dim3(C_ / 16, H_ / 16, B_), 32, 0, stream>>>(QT, hT, out);
  k_aq<<<dim3(N_ / 16, C_ / 16, B_), 32, 0, stream>>>(adj8, QT, T1T);
  k_apool<<<dim3(C_ / 16, C_ / 16, B_), 32, 0, stream>>>(QT, T1T, out + (size_t)B_ * C_ * H_);
}
